// SelectiveSSM_2894807958203
// MI455X (gfx1250) — matrix operand order repaired and checked by reading, not compiled, not run on hardware
//
#include <hip/hip_runtime.h>
#include <hip/hip_bf16.h>
#include <math.h>

// Problem constants (match reference)
#define BB      2
#define SEQ     2048
#define DM      1024
#define NST     16
#define RNK     64
#define MTOT    (BB * SEQ)          // 4096 rows for all GEMMs

typedef __bf16  bf16_t;
typedef __attribute__((ext_vector_type(16))) __bf16 v16bf;
typedef __attribute__((ext_vector_type(8)))  __bf16 v8bf;
typedef __attribute__((ext_vector_type(8)))  float  v8f;

// ---------------------------------------------------------------------------
// Kernel 1: causal depthwise conv1d (k=4) + bias + SiLU.
// Writes xc in f32 (for the scan / residual) and bf16 (WMMA A-operand).
// ---------------------------------------------------------------------------
__global__ void conv_silu_kernel(const float* __restrict__ x,
                                 const float* __restrict__ cw,
                                 const float* __restrict__ cb,
                                 float* __restrict__ xc,
                                 bf16_t* __restrict__ xcb) {
    int idx = blockIdx.x * blockDim.x + threadIdx.x;   // over B*L*D
    int d = idx & (DM - 1);
    int l = (idx / DM) & (SEQ - 1);
    int b = idx / (DM * SEQ);
    const float w0 = cw[d * 4 + 0], w1 = cw[d * 4 + 1];
    const float w2 = cw[d * 4 + 2], w3 = cw[d * 4 + 3];
    const float* xb = x + (size_t)b * SEQ * DM + d;
    float acc = cb[d];
    if (l >= 3) acc += w0 * xb[(size_t)(l - 3) * DM];
    if (l >= 2) acc += w1 * xb[(size_t)(l - 2) * DM];
    if (l >= 1) acc += w2 * xb[(size_t)(l - 1) * DM];
    acc += w3 * xb[(size_t)l * DM];
    float s = acc / (1.0f + __expf(-acc));             // SiLU
    xc[idx]  = s;
    xcb[idx] = (bf16_t)s;
}

// ---------------------------------------------------------------------------
// Kernel 2: generic f32 -> bf16 convert (weights).
// ---------------------------------------------------------------------------
__global__ void cvt_bf16_kernel(const float* __restrict__ src,
                                bf16_t* __restrict__ dst, int n) {
    int i = blockIdx.x * blockDim.x + threadIdx.x;
    if (i < n) dst[i] = (bf16_t)src[i];
}

// ---------------------------------------------------------------------------
// Kernel 3: bf16 WMMA GEMM   out[M,N] = A[M,K] * W[N,K]^T   (one wave / 16x16 tile)
// mode 0: store f32     mode 1: store bf16     mode 2: +bias, softplus, store f32
// A and B fragments both use contiguous k order: element s of lane half h is k = 16h + s.
// ---------------------------------------------------------------------------
__global__ void gemm_wmma_bf16(const bf16_t* __restrict__ A,
                               const bf16_t* __restrict__ W,
                               int M, int N, int K,
                               float* __restrict__ outF,
                               bf16_t* __restrict__ outB,
                               const float* __restrict__ bias,
                               int mode) {
    const int lane    = threadIdx.x & 31;
    const int wave    = threadIdx.x >> 5;
    const int tiles_n = N >> 4;
    const int tile    = blockIdx.x * (blockDim.x >> 5) + wave;
    const int tile_m  = tile / tiles_n;
    const int tile_n  = tile - tile_m * tiles_n;
    const int half    = lane >> 4;                     // 0: lanes 0-15, 1: lanes 16-31
    const int l15     = lane & 15;

    // A fragment: row (tile_m*16 + l15), two contiguous 8-half runs at
    // k = kk + half*16 and k = kk + half*16 + 8.
    const bf16_t* Ap = A + (size_t)(tile_m * 16 + l15) * K + half * 16;
    // B fragment: weight row (tile_n*16 + l15), contiguous 16 halfs at
    // k = kk + half*16.
    const bf16_t* Wp = W + (size_t)(tile_n * 16 + l15) * K + half * 16;

    v8f c = {};
    for (int kk = 0; kk < K; kk += 32) {
        v8bf alo = *(const v8bf*)(Ap + kk);
        v8bf ahi = *(const v8bf*)(Ap + kk + 8);
        v16bf a, bfr;
        #pragma unroll
        for (int i = 0; i < 8; ++i) { a[i] = alo[i]; a[i + 8] = ahi[i]; }
        bfr = *(const v16bf*)(Wp + kk);
        // (neg_a, A, neg_b, B, c_mod, C, reuse_a, reuse_b)
        c = __builtin_amdgcn_wmma_f32_16x16x32_bf16(
                false, a, false, bfr, (short)0, c, false, false);
    }

    // D layout: element r -> m = tile_m*16 + half*8 + r, n = tile_n*16 + l15
    const int ncol  = tile_n * 16 + l15;
    const int mbase = tile_m * 16 + half * 8;
    if (mode == 2) {
        const float bv = bias[ncol];
        #pragma unroll
        for (int r = 0; r < 8; ++r) {
            float z = c[r] + bv;
            z = (z > 20.0f) ? z : log1pf(__expf(z));   // softplus
            outF[(size_t)(mbase + r) * N + ncol] = z;
        }
    } else if (mode == 1) {
        #pragma unroll
        for (int r = 0; r < 8; ++r)
            outB[(size_t)(mbase + r) * N + ncol] = (bf16_t)c[r];
    } else {
        #pragma unroll
        for (int r = 0; r < 8; ++r)
            outF[(size_t)(mbase + r) * N + ncol] = c[r];
    }
}

// ---------------------------------------------------------------------------
// Kernel 4: selective-scan recurrence. One thread per (b, d) channel,
// 16-element state in VGPRs. Shared B/C (32 floats per timestep) staged
// through LDS in 16-step chunks; next chunk prefetched (global_prefetch_b8).
// ---------------------------------------------------------------------------
#define TCH 16
__global__ void scan_kernel(const float* __restrict__ xc,
                            const float* __restrict__ dt,
                            const float* __restrict__ BC,    // (B, L, 32): B then C
                            const float* __restrict__ A_log,
                            const float* __restrict__ Dp,
                            float* __restrict__ out) {
    __shared__ float sBC[TCH * 32];
    const int b = blockIdx.x >> 2;                    // DM/256 = 4 blocks per batch
    const int d = ((blockIdx.x & 3) << 8) + threadIdx.x;

    float Arow[NST], h[NST];
    #pragma unroll
    for (int n = 0; n < NST; ++n) {
        Arow[n] = -__expf(A_log[d * NST + n]);
        h[n] = 0.0f;
    }
    const float dval = Dp[d];
    const float* bcb = BC + (size_t)b * SEQ * 32;

    for (int t0 = 0; t0 < SEQ; t0 += TCH) {
        __syncthreads();                               // prior chunk fully consumed
        {
            int f = threadIdx.x;                       // 512 floats, 2 per thread
            sBC[f]       = bcb[(size_t)t0 * 32 + f];
            sBC[f + 256] = bcb[(size_t)t0 * 32 + f + 256];
        }
        __syncthreads();
        if (t0 + TCH < SEQ)                            // warm L2/L0 for next chunk
            __builtin_prefetch(&bcb[(size_t)(t0 + TCH) * 32 + threadIdx.x * 2], 0, 3);

        for (int tt = 0; tt < TCH; ++tt) {
            const size_t idx = ((size_t)b * SEQ + (t0 + tt)) * DM + d;
            const float dtv = dt[idx];
            const float xv  = xc[idx];
            const float dbx = dtv * xv;
            float y = 0.0f;
            #pragma unroll
            for (int n = 0; n < NST; ++n) {
                float a = __expf(dtv * Arow[n]);       // v_exp_f32 (TRANS)
                h[n] = a * h[n] + dbx * sBC[tt * 32 + n];
                y += h[n] * sBC[tt * 32 + 16 + n];
            }
            out[idx] = y + dval * xv;
        }
    }
}

// ---------------------------------------------------------------------------
extern "C" void kernel_launch(void* const* d_in, const int* in_sizes, int n_in,
                              void* d_out, int out_size, void* d_ws, size_t ws_size,
                              hipStream_t stream) {
    (void)in_sizes; (void)n_in; (void)out_size; (void)ws_size;

    const float* x      = (const float*)d_in[0];
    const float* conv_w = (const float*)d_in[1];
    const float* conv_b = (const float*)d_in[2];
    const float* Wxdt   = (const float*)d_in[3];   // (64, 1024)
    const float* Wdt    = (const float*)d_in[4];   // (1024, 64)
    const float* bdt    = (const float*)d_in[5];   // (1024,)
    const float* WB     = (const float*)d_in[6];   // (16, 1024)
    const float* WC     = (const float*)d_in[7];   // (16, 1024)
    const float* A_log  = (const float*)d_in[8];   // (1024, 16)
    const float* Dp     = (const float*)d_in[9];   // (1024,)
    float* out = (float*)d_out;

    // workspace carve-up (all offsets >= 32B aligned for b128/b256 loads)
    char* ws = (char*)d_ws;
    float*  xc_f   = (float*)(ws);                              // 16 MB
    float*  dt_f   = (float*)(ws + 16777216);                   // 16 MB
    float*  bc_f   = (float*)(ws + 33554432);                   // 512 KB
    bf16_t* xc_b   = (bf16_t*)(ws + 34078720);                  // 8 MB
    bf16_t* dtl_b  = (bf16_t*)(ws + 42467328);                  // 512 KB
    bf16_t* wx_b   = (bf16_t*)(ws + 42991616);                  // 128 KB
    bf16_t* wdt_b  = (bf16_t*)(ws + 43122688);                  // 128 KB
    bf16_t* wbc_b  = (bf16_t*)(ws + 43253760);                  // 64 KB (B rows 0-15, C rows 16-31)

    // 1) conv + SiLU
    conv_silu_kernel<<<(BB * SEQ * DM) / 256, 256, 0, stream>>>(
        x, conv_w, conv_b, xc_f, xc_b);

    // 2) weight converts to bf16
    cvt_bf16_kernel<<<(RNK * DM) / 256, 256, 0, stream>>>(Wxdt, wx_b, RNK * DM);
    cvt_bf16_kernel<<<(DM * RNK) / 256, 256, 0, stream>>>(Wdt,  wdt_b, DM * RNK);
    cvt_bf16_kernel<<<(NST * DM) / 256, 256, 0, stream>>>(WB, wbc_b,            NST * DM);
    cvt_bf16_kernel<<<(NST * DM) / 256, 256, 0, stream>>>(WC, wbc_b + NST * DM, NST * DM);

    // 3) dt_low[4096,64] = xc * Wxdt^T      (K=1024) -> bf16
    {
        int tiles = (MTOT / 16) * (RNK / 16);                   // 1024
        gemm_wmma_bf16<<<tiles / 8, 256, 0, stream>>>(
            xc_b, wx_b, MTOT, RNK, DM, nullptr, dtl_b, nullptr, 1);
    }
    // 4) dt[4096,1024] = softplus(dt_low * Wdt^T + b)   (K=64)
    {
        int tiles = (MTOT / 16) * (DM / 16);                    // 16384
        gemm_wmma_bf16<<<tiles / 8, 256, 0, stream>>>(
            dtl_b, wdt_b, MTOT, DM, RNK, dt_f, nullptr, bdt, 2);
    }
    // 5) BC[4096,32] = xc * [WB;WC]^T       (K=1024)
    {
        int tiles = (MTOT / 16) * (32 / 16);                    // 512
        gemm_wmma_bf16<<<tiles / 8, 256, 0, stream>>>(
            xc_b, wbc_b, MTOT, 32, DM, bc_f, nullptr, nullptr, 0);
    }
    // 6) recurrent selective scan + residual
    scan_kernel<<<BB * (DM / 256), 256, 0, stream>>>(
        xc_f, dt_f, bc_f, A_log, Dp, out);
}
